// liGRU_44925357916490
// MI455X (gfx1250) — matrix operand order repaired and checked by reading, not compiled, not run on hardware
//
#include <hip/hip_runtime.h>
#include <hip/hip_bf16.h>

typedef unsigned short u16;
typedef __attribute__((ext_vector_type(16))) __bf16 v16bf;
typedef __attribute__((ext_vector_type(8)))  float  v8f;
typedef __attribute__((ext_vector_type(8)))  int    v8i;

// ---------------------------------------------------------------- helpers ---

__device__ __forceinline__ u16 f2bf(float f) {
    unsigned u = __builtin_bit_cast(unsigned, f);
    unsigned rnd = 0x7FFFu + ((u >> 16) & 1u);   // round-to-nearest-even
    return (u16)((u + rnd) >> 16);
}

__device__ __forceinline__ v16bf pack32B(int4 lo, int4 hi) {
    v8i t;
    t[0] = lo.x; t[1] = lo.y; t[2] = lo.z; t[3] = lo.w;
    t[4] = hi.x; t[5] = hi.y; t[6] = hi.z; t[7] = hi.w;
    return __builtin_bit_cast(v16bf, t);
}

// A fragment: 16x32 bf16, row-major src [*, ld], tile at (m0, k0).
// lanes 0-15: row M=lane, K = k0+0..7 (v0..3) and k0+16..23 (v4..7)
// lanes16-31: row M=lane-16, K = k0+8..15 and k0+24..31
__device__ __forceinline__ v16bf load_A(const u16* src, int ld, int m0, int k0, int lane) {
    int r  = m0 + (lane & 15);
    int kb = k0 + ((lane >> 4) << 3);
    const u16* p = src + (long)r * ld + kb;
    int4 lo = *(const int4*)(p);
    int4 hi = *(const int4*)(p + 16);
    return pack32B(lo, hi);
}

// B fragment: 32x16 bf16 where B[k,n] = W[n,k], W row-major [*, ld], cols n0..n0+15.
// lanes 0-15: col N=lane, K = k0+0..7 and k0+16..23
// lanes16-31: col N=lane-16, K = k0+8..15 and k0+24..31
__device__ __forceinline__ v16bf load_B(const u16* W, int ld, int n0, int k0, int lane) {
    int r  = n0 + (lane & 15);
    int kb = k0 + ((lane >> 4) << 3);
    const u16* p = W + (long)r * ld + kb;
    int4 lo = *(const int4*)(p);
    int4 hi = *(const int4*)(p + 16);
    return pack32B(lo, hi);
}

__device__ __forceinline__ v8f wmma_bf16(v16bf a, v16bf b, v8f c) {
    return __builtin_amdgcn_wmma_f32_16x16x32_bf16(
        /*neg_a=*/false, a, /*neg_b=*/false, b,
        /*c_mod=*/(short)0, c, /*reuse_a=*/false, /*reuse_b=*/false);
}

// CDNA5 async copy: 16B global -> LDS, tracked with ASYNCcnt (no VGPR return).
__device__ __forceinline__ void async_g2l_b128(unsigned lds_off, const void* gptr) {
    asm volatile("global_load_async_to_lds_b128 %0, %1, off"
                 :: "v"(lds_off), "v"((unsigned long long)(size_t)gptr)
                 : "memory");
}
__device__ __forceinline__ void wait_async0() {
    asm volatile("s_wait_asynccnt 0x0" ::: "memory");
}

// -------------------------------------------------------- convert kernels ---

__global__ __launch_bounds__(256) void f32_to_bf16_vec(
        const float* __restrict__ src, u16* __restrict__ dst, long n4) {
    long i = (long)blockIdx.x * blockDim.x + threadIdx.x;
    if (i >= n4) return;
    float4 v = ((const float4*)src)[i];
    unsigned lo = (unsigned)f2bf(v.x) | ((unsigned)f2bf(v.y) << 16);
    unsigned hi = (unsigned)f2bf(v.z) | ((unsigned)f2bf(v.w) << 16);
    ((uint2*)dst)[i] = make_uint2(lo, hi);
}

// ----------------------------------------------------- input GEMM (WMMA) ----
// U[M,N] = Xbf[M,K] * Wbf[N,K]^T + bias[N]   (f32 accumulate/output)
// 256 threads = 8 wave32. WG tile 128x128, k-slice 32, double-buffered LDS
// fed by global_load_async_to_lds_b128. Wave tile 32(M) x 64(N) = 8 acc.

#define GEMM_BM  128
#define GEMM_BN  128
#define GEMM_KB  32
#define GEMM_LDT 40          // padded LDS row (halves): 80B, bank-conflict-free

__global__ __launch_bounds__(256) void gemm_bias_wmma(
        const u16* __restrict__ X, const u16* __restrict__ W,
        const float* __restrict__ bias, float* __restrict__ U,
        int M, int N, int K) {
    extern __shared__ u16 smem[];
    u16* Abuf = smem;                               // [2][BM][LDT]
    u16* Bbuf = smem + 2 * GEMM_BM * GEMM_LDT;      // [2][BN][LDT]

    const int tid   = threadIdx.x;
    const int wave  = tid >> 5;
    const int lane  = tid & 31;
    const int mBase = blockIdx.y * GEMM_BM;
    const int nBase = blockIdx.x * GEMM_BN;
    const int wm = (wave >> 1) * 32;   // wave M offset in WG tile
    const int wn = (wave & 1) * 64;    // wave N offset in WG tile

    v8f acc[2][4];
#pragma unroll
    for (int i = 0; i < 2; ++i)
#pragma unroll
        for (int j = 0; j < 4; ++j)
            acc[i][j] = (v8f){0.f, 0.f, 0.f, 0.f, 0.f, 0.f, 0.f, 0.f};

    // issue one k-slice: 128 rows x 64B for A and B = 512 16B-chunks each;
    // 256 threads -> 2 A-chunks + 2 B-chunks per thread.
    auto issue = [&](int buf, int k0) {
#pragma unroll
        for (int s = 0; s < 2; ++s) {
            int c   = tid + s * 256;       // chunk id 0..511
            int row = c >> 2;
            int kc  = (c & 3) * 8;         // halves
            const u16* gA = X + (size_t)(mBase + row) * K + k0 + kc;
            unsigned   lA = (unsigned)(size_t)(Abuf +
                              ((size_t)buf * GEMM_BM + row) * GEMM_LDT + kc);
            async_g2l_b128(lA, gA);
            const u16* gB = W + (size_t)(nBase + row) * K + k0 + kc;
            unsigned   lB = (unsigned)(size_t)(Bbuf +
                              ((size_t)buf * GEMM_BN + row) * GEMM_LDT + kc);
            async_g2l_b128(lB, gB);
        }
    };

    issue(0, 0);
    const int KT = K / GEMM_KB;
    for (int kt = 0; kt < KT; ++kt) {
        int cur = kt & 1;
        wait_async0();          // this wave's in-flight slice landed
        __syncthreads();        // every wave's slice landed
        if (kt + 1 < KT) issue(cur ^ 1, (kt + 1) * GEMM_KB);  // overlap next

        const u16* At = Abuf + (size_t)cur * GEMM_BM * GEMM_LDT;
        const u16* Bt = Bbuf + (size_t)cur * GEMM_BN * GEMM_LDT;
        v16bf af[2], bfr[4];
#pragma unroll
        for (int i = 0; i < 2; ++i)
            af[i] = load_A(At, GEMM_LDT, wm + i * 16, 0, lane);
#pragma unroll
        for (int j = 0; j < 4; ++j)
            bfr[j] = load_B(Bt, GEMM_LDT, wn + j * 16, 0, lane);
#pragma unroll
        for (int i = 0; i < 2; ++i)
#pragma unroll
            for (int j = 0; j < 4; ++j)
                acc[i][j] = wmma_bf16(af[i], bfr[j], acc[i][j]);
        // next iteration's wait_async0 + __syncthreads fences buf reuse
    }

    const int col  = lane & 15;
    const int row0 = (lane >> 4) * 8;
#pragma unroll
    for (int i = 0; i < 2; ++i) {
#pragma unroll
        for (int j = 0; j < 4; ++j) {
            int n = nBase + wn + j * 16 + col;
            float bv = bias[n];
            float* outp = U + (size_t)(mBase + wm + i * 16 + row0) * N + n;
#pragma unroll
            for (int r = 0; r < 8; ++r)
                outp[(size_t)r * N] = acc[i][j][r] + bv;
        }
    }
}

// --------------------------------------------- BatchNorm over batch axis ----
// In-place on U[T,B,H]; one thread per (t, h); biased variance, eps=1e-5.

__global__ __launch_bounds__(256) void bn_kernel(
        float* __restrict__ U, const float* __restrict__ gamma,
        const float* __restrict__ beta, int T, int B, int H) {
    long idx = (long)blockIdx.x * blockDim.x + threadIdx.x;
    if (idx >= (long)T * H) return;
    int t = (int)(idx / H);
    int n = (int)(idx % H);
    float* p = U + (long)t * B * H + n;
    float v[32];
    float s = 0.f, s2 = 0.f;
#pragma unroll
    for (int i = 0; i < 32; ++i) {
        v[i] = p[(long)i * H];
        s  += v[i];
        s2 += v[i] * v[i];
    }
    float mean = s * (1.f / 32.f);
    float var  = s2 * (1.f / 32.f) - mean * mean;
    float sc = rsqrtf(var + 1e-5f) * gamma[n];
    float bb = beta[n];
#pragma unroll
    for (int i = 0; i < 32; ++i)
        p[(long)i * H] = (v[i] - mean) * sc + bb;
}

// ----------------------------------------- persistent recurrence (WMMA) -----
// 32 WGs; WG g owns output rows [32g, 32g+32). Recurrent weights for the
// slice live in LDS (bf16, 128KB). Per step: 8 waves compute 4 batch x slice
// 16x16 tiles for each of the two gates (K=1024 -> 32 WMMAs/wave/step),
// stage pre-activations in LDS, fuse gate math elementwise, publish h_{t+1}
// (f32 + bf16, ping-pong buffers), then grid-barrier on bar[t].

__global__ __launch_bounds__(256) void ligru_recurrence(
        const float* __restrict__ Uz, const float* __restrict__ Uh,
        const float* __restrict__ bhz, const float* __restrict__ bhh,
        const u16* __restrict__ Whz, const u16* __restrict__ Whh,   // [H,H] bf16
        float* __restrict__ hf, u16* __restrict__ hb,               // [2][B*H]
        u16* __restrict__ xnext, float* __restrict__ yout,
        float* __restrict__ hlast,
        unsigned* __restrict__ bar, int T) {
    const int H = 1024, B = 32;
    extern __shared__ u16 lds[];
    u16* Wz_s = lds;                 // 32*1024 bf16 = 64KB
    u16* Wh_s = lds + 32 * 1024;     // 64KB
    float* zbuf = (float*)(lds + 64 * 1024);  // 32x32 f32 staging
    float* cbuf = zbuf + 1024;

    int n0 = blockIdx.x * 32;
    // load weight slices into LDS via async engine (16B granularity)
    {
        const u16* gz = Whz + (long)n0 * H;
        const u16* gh = Whh + (long)n0 * H;
        for (int i = threadIdx.x; i < 32 * 1024 / 8; i += blockDim.x) {
            async_g2l_b128((unsigned)(size_t)(Wz_s + i * 8), gz + i * 8);
            async_g2l_b128((unsigned)(size_t)(Wh_s + i * 8), gh + i * 8);
        }
        wait_async0();
    }
    __syncthreads();

    int wave = threadIdx.x >> 5;
    int lane = threadIdx.x & 31;
    int gate = wave >> 2;                 // 0 = z gate, 1 = candidate
    int mt = ((wave >> 1) & 1) * 16;      // batch tile
    int nt = (wave & 1) * 16;             // slice-column tile
    const u16* Ws = gate ? Wh_s : Wz_s;
    float* sbuf = gate ? cbuf : zbuf;
    const unsigned nwg = gridDim.x;

    for (int t = 0; t < T; ++t) {
        const u16*   hcur   = hb + (t & 1) * (B * H);
        const float* hcur_f = hf + (t & 1) * (B * H);
        float*       hnxt_f = hf + ((t + 1) & 1) * (B * H);
        u16*         hnxt_b = hb + ((t + 1) & 1) * (B * H);

        v8f acc = {0.f, 0.f, 0.f, 0.f, 0.f, 0.f, 0.f, 0.f};
#pragma unroll 4
        for (int k0 = 0; k0 < H; k0 += 32) {
            v16bf a = load_A(hcur, H, mt, k0, lane);   // global (L2-hot)
            v16bf b = load_B(Ws, H, nt, k0, lane);     // LDS
            acc = wmma_bf16(a, b, acc);
        }
        int col  = lane & 15;
        int row0 = (lane >> 4) * 8;
#pragma unroll
        for (int r = 0; r < 8; ++r)
            sbuf[(mt + row0 + r) * 32 + nt + col] = acc[r];
        __syncthreads();

        // fuse gates: 1024 elements (32 batch x 32 slice cols)
        for (int i = threadIdx.x; i < B * 32; i += blockDim.x) {
            int b_ = i >> 5;
            int nn = i & 31;
            int n  = n0 + nn;
            long off = (long)t * B * H + (long)b_ * H + n;
            float zp = Uz[off] + zbuf[i] + bhz[n];
            float cp = Uh[off] + cbuf[i] + bhh[n];
            float z    = 1.f / (1.f + __expf(-zp));
            float cand = fmaxf(cp, 0.f);
            float hprev = hcur_f[(long)b_ * H + n];
            float hnew  = z * hprev + (1.f - z) * cand;
            long hoff = (long)b_ * H + n;
            hnxt_f[hoff] = hnew;
            hnxt_b[hoff] = f2bf(hnew);
            if (xnext) xnext[off] = f2bf(hnew);       // next layer's bf16 input
            if (yout)  yout[off]  = hnew;             // final-layer f32 output
            if (t == T - 1) hlast[hoff] = hnew;       // final hidden state
        }
        __syncthreads();          // staging consumed before next overwrite
        __threadfence();          // h_{t+1} visible device-wide
        if (threadIdx.x == 0) {
            __hip_atomic_fetch_add(&bar[t], 1u, __ATOMIC_ACQ_REL,
                                   __HIP_MEMORY_SCOPE_AGENT);
            while (__hip_atomic_load(&bar[t], __ATOMIC_ACQUIRE,
                                     __HIP_MEMORY_SCOPE_AGENT) < nwg)
                __builtin_amdgcn_s_sleep(1);
        }
        __syncthreads();
    }
}

// ------------------------------------------------------------------ host ----

extern "C" void kernel_launch(void* const* d_in, const int* in_sizes, int n_in,
                              void* d_out, int out_size, void* d_ws, size_t ws_size,
                              hipStream_t stream) {
    const int T = 512, B = 32, I = 1024, H = 1024, L = 2;
    const long M = (long)T * B;            // 16384 GEMM rows

    const float* x   = (const float*)d_in[0];
    const float* Wxz = (const float*)d_in[2];
    const float* bxz = (const float*)d_in[3];
    const float* gz  = (const float*)d_in[4];
    const float* bz  = (const float*)d_in[5];
    const float* Whz = (const float*)d_in[6];
    const float* bhz = (const float*)d_in[7];
    const float* Wxh = (const float*)d_in[8];
    const float* bxh = (const float*)d_in[9];
    const float* gh  = (const float*)d_in[10];
    const float* bh  = (const float*)d_in[11];
    const float* Whh = (const float*)d_in[12];
    const float* bhh = (const float*)d_in[13];
    float* out = (float*)d_out;            // [T,B,H] ++ [L,B,H]

    char* ws = (char*)d_ws;
    size_t off = 0;
    auto alloc = [&](size_t bytes) -> char* {
        char* p = ws + off;
        off += (bytes + 255) & ~(size_t)255;
        return p;
    };
    u16*      Xbf  = (u16*)     alloc(M * I * sizeof(u16));          // 32MB
    float*    Uz   = (float*)   alloc(M * H * sizeof(float));        // 64MB
    float*    Uh   = (float*)   alloc(M * H * sizeof(float));        // 64MB
    u16*      WxzB = (u16*)     alloc((size_t)L * H * I * sizeof(u16));
    u16*      WxhB = (u16*)     alloc((size_t)L * H * I * sizeof(u16));
    u16*      WhzB = (u16*)     alloc((size_t)L * H * H * sizeof(u16));
    u16*      WhhB = (u16*)     alloc((size_t)L * H * H * sizeof(u16));
    float*    hf   = (float*)   alloc(2 * B * H * sizeof(float));
    u16*      hb   = (u16*)     alloc(2 * B * H * sizeof(u16));
    unsigned* bar  = (unsigned*)alloc((size_t)L * T * sizeof(unsigned));

    // one-shot conversions (graph-captured nodes, rerun each replay)
    {
        long n4 = M * I / 4;
        f32_to_bf16_vec<<<(unsigned)((n4 + 255) / 256), 256, 0, stream>>>(x, Xbf, n4);
        long w4 = (long)L * H * I / 4;
        f32_to_bf16_vec<<<(unsigned)((w4 + 255) / 256), 256, 0, stream>>>(Wxz, WxzB, w4);
        f32_to_bf16_vec<<<(unsigned)((w4 + 255) / 256), 256, 0, stream>>>(Wxh, WxhB, w4);
        f32_to_bf16_vec<<<(unsigned)((w4 + 255) / 256), 256, 0, stream>>>(Whz, WhzB, w4);
        f32_to_bf16_vec<<<(unsigned)((w4 + 255) / 256), 256, 0, stream>>>(Whh, WhhB, w4);
    }
    hipMemsetAsync(bar, 0, (size_t)L * T * sizeof(unsigned), stream);

    const size_t gemmLds = (size_t)2 * (GEMM_BM + GEMM_BN) * GEMM_LDT * sizeof(u16); // 40KB
    const size_t recLds  = (64 + 64 + 4 + 4) * 1024;                                 // 136KB

    for (int l = 0; l < L; ++l) {
        // h0 = 0 for every layer
        hipMemsetAsync(hf, 0, 2 * B * H * sizeof(float), stream);
        hipMemsetAsync(hb, 0, 2 * B * H * sizeof(u16), stream);

        // input-side GEMMs (compute-bound, bf16 WMMA, async-LDS pipelined)
        dim3 gg(H / GEMM_BN, (unsigned)(M / GEMM_BM));
        gemm_bias_wmma<<<gg, 256, gemmLds, stream>>>(
            Xbf, WxzB + (size_t)l * H * I, bxz + (size_t)l * H, Uz,
            (int)M, H, I);
        gemm_bias_wmma<<<gg, 256, gemmLds, stream>>>(
            Xbf, WxhB + (size_t)l * H * I, bxh + (size_t)l * H, Uh,
            (int)M, H, I);

        // per-timestep BatchNorm over the batch axis
        long nTH = (long)T * H;
        bn_kernel<<<(unsigned)((nTH + 255) / 256), 256, 0, stream>>>(
            Uz, gz + (size_t)l * H, bz + (size_t)l * H, T, B, H);
        bn_kernel<<<(unsigned)((nTH + 255) / 256), 256, 0, stream>>>(
            Uh, gh + (size_t)l * H, bh + (size_t)l * H, T, B, H);

        // sequential scan: persistent kernel, LDS-resident recurrent weights
        u16*   xnext = (l == L - 1) ? nullptr : Xbf;   // overwrite in place
        float* yout  = (l == L - 1) ? out : nullptr;
        float* hlast = out + (size_t)T * B * H + (size_t)l * B * H;
        ligru_recurrence<<<32, 256, recLds, stream>>>(
            Uz, Uh, bhz + (size_t)l * H, bhh + (size_t)l * H,
            WhzB + (size_t)l * H * H, WhhB + (size_t)l * H * H,
            hf, hb, xnext, yout, hlast, bar + (size_t)l * T, T);
    }
}
